// CausalMultiHeadSelfAttention_12446815224598
// MI455X (gfx1250) — compile-verified, matrix operand order checked by reading, not run on hardware
//
#include <hip/hip_runtime.h>

// ---------------------------------------------------------------------------
// CausalMultiHeadSelfAttention for MI455X (gfx1250, wave32, WMMA).
// B=4, S=2048, D_MODEL=2048, H=16, Dk=128, fp32 in/out, RoPE theta=1e4.
// bf16 data path + f32 WMMA accumulation (v_wmma_f32_16x16x32_bf16),
// flash-attention with online softmax, V stored transposed so every matmul is
// the same A(MxK) * B(NxK)^T WMMA pattern. GEMM tiles are staged into LDS with
// gfx1250 async-to-LDS loads (double buffered) when the toolchain exposes them.
// ---------------------------------------------------------------------------

typedef __attribute__((ext_vector_type(16))) __bf16 bf16x16;
typedef __attribute__((ext_vector_type(8)))  __bf16 bf16x8;
typedef __attribute__((ext_vector_type(4)))  __bf16 bf16x4;
typedef __attribute__((ext_vector_type(8)))  float  f32x8;
typedef __attribute__((ext_vector_type(4)))  int    i32x4;

#define S_LEN   2048
#define DMODEL  2048
#define NHEADS  16
#define DK      128
#define NBATCH  4

#if defined(__HIP_DEVICE_COMPILE__) && __has_builtin(__builtin_amdgcn_global_load_async_to_lds_b128)
#define USE_ASYNC_LDS 1
typedef __attribute__((address_space(1))) i32x4* gp128;   // global int4*
typedef __attribute__((address_space(3))) i32x4* lp128;   // LDS int4*
#else
#define USE_ASYNC_LDS 0
#endif

template <int N>
__device__ __forceinline__ void wait_async() {
#if USE_ASYNC_LDS
#if __has_builtin(__builtin_amdgcn_s_wait_asynccnt)
    __builtin_amdgcn_s_wait_asynccnt(N);
#else
    asm volatile("s_wait_asynccnt %0" ::"n"(N) : "memory");
#endif
#endif
}

__device__ __forceinline__ f32x8 wmma_bf16(bf16x16 a, bf16x16 b, f32x8 c) {
    // D = A(16x32 bf16) * B(32x16 bf16, given as N x K frag) + C(16x16 f32)
    return __builtin_amdgcn_wmma_f32_16x16x32_bf16(
        /*neg_a=*/false, a, /*neg_b=*/false, b,
        /*c_mod=*/(short)0, c, /*reuse_a=*/false, /*reuse_b=*/false);
}

// Load a 16x32 bf16 fragment (A-layout; the B operand uses the same layout
// with "rows" = N). base points at (row0, col0); stride in elements.
__device__ __forceinline__ bf16x16 load_frag(const __bf16* base, int stride, int lane) {
    const int r  = lane & 15;
    const int hi = lane >> 4;
    const __bf16* p = base + (size_t)r * stride + hi * 8;
    bf16x8 lo = *(const bf16x8*)p;          // K = hi*8 .. +7
    bf16x8 hp = *(const bf16x8*)(p + 16);   // K = 16 + hi*8 .. +7
    bf16x16 out;
#pragma unroll
    for (int i = 0; i < 8; ++i) { out[i] = lo[i]; out[i + 8] = hp[i]; }
    return out;
}

// ---------------------------------------------------------------------------
// fp32 -> bf16 conversion (vectorized x4; all sizes are multiples of 4)
// ---------------------------------------------------------------------------
__global__ void cvt_f32_bf16_kernel(const float* __restrict__ in,
                                    __bf16* __restrict__ out, int n4) {
    int i = blockIdx.x * blockDim.x + threadIdx.x;
    const int stride = gridDim.x * blockDim.x;
    const float4* in4 = (const float4*)in;
    bf16x4* out4 = (bf16x4*)out;
    for (; i < n4; i += stride) {
        float4 f = in4[i];
        bf16x4 o;
        o[0] = (__bf16)f.x; o[1] = (__bf16)f.y;
        o[2] = (__bf16)f.z; o[3] = (__bf16)f.w;
        out4[i] = o;
    }
}

// ---------------------------------------------------------------------------
// Tiled GEMM: C[M x N] = A[M x K] * B[N x K]^T, bf16 in, f32 accumulate.
// Block = 256 threads (8 waves), tile 128x128, K-step 32, double-buffered
// async LDS staging.
// MODE 0: store f32 C (final output projection)
// MODE 1: fuse RoPE, store bf16 to [B][H][S][Dk]   (Q and K)
// MODE 2: store bf16 transposed to [B][H][Dk][S]   (V)
// ---------------------------------------------------------------------------
#define LDSB 40   // padded LDS row stride (halfwords): 80B, 16B aligned, conflict-free

template <int MODE>
__global__ __launch_bounds__(256)
void gemm_bf16_abt_kernel(const __bf16* __restrict__ A, int lda,
                          const __bf16* __restrict__ Bm, int ldb,
                          float* __restrict__ Cf, int ldc,
                          __bf16* __restrict__ Obf,
                          const int* __restrict__ tpos,
                          int K) {
    __shared__ __bf16 As[2][128 * LDSB];
    __shared__ __bf16 Bs[2][128 * LDSB];

    const int t    = threadIdx.x;
    const int lane = t & 31;
    const int wave = t >> 5;
    const int wm   = wave >> 1;   // 0..3 : 32-row strip
    const int wn   = wave & 1;    // 0..1 : 64-col strip
    const int m0   = blockIdx.y * 128;
    const int n0   = blockIdx.x * 128;

    // this thread's two 16B chunks of the 128x32 tile (512 chunks / 256 thr)
    const int row0 = t >> 2;
    const int kc0  = (t & 3) * 8;
    const int row1 = (t + 256) >> 2;
    const int kc1  = ((t + 256) & 3) * 8;

    f32x8 acc[2][4];
#pragma unroll
    for (int mf = 0; mf < 2; ++mf)
#pragma unroll
        for (int nf = 0; nf < 4; ++nf) acc[mf][nf] = (f32x8){};

    // stage one 128x32 K-slab of A and B into LDS buffer `buf`
    auto stage = [&](int k0, int buf) {
#if USE_ASYNC_LDS
        __builtin_amdgcn_global_load_async_to_lds_b128(
            (gp128)(A + (size_t)(m0 + row0) * lda + k0 + kc0),
            (lp128)&As[buf][row0 * LDSB + kc0], 0, 0);
        __builtin_amdgcn_global_load_async_to_lds_b128(
            (gp128)(A + (size_t)(m0 + row1) * lda + k0 + kc1),
            (lp128)&As[buf][row1 * LDSB + kc1], 0, 0);
        __builtin_amdgcn_global_load_async_to_lds_b128(
            (gp128)(Bm + (size_t)(n0 + row0) * ldb + k0 + kc0),
            (lp128)&Bs[buf][row0 * LDSB + kc0], 0, 0);
        __builtin_amdgcn_global_load_async_to_lds_b128(
            (gp128)(Bm + (size_t)(n0 + row1) * ldb + k0 + kc1),
            (lp128)&Bs[buf][row1 * LDSB + kc1], 0, 0);
#else
        *(bf16x8*)&As[buf][row0 * LDSB + kc0] =
            *(const bf16x8*)(A + (size_t)(m0 + row0) * lda + k0 + kc0);
        *(bf16x8*)&As[buf][row1 * LDSB + kc1] =
            *(const bf16x8*)(A + (size_t)(m0 + row1) * lda + k0 + kc1);
        *(bf16x8*)&Bs[buf][row0 * LDSB + kc0] =
            *(const bf16x8*)(Bm + (size_t)(n0 + row0) * ldb + k0 + kc0);
        *(bf16x8*)&Bs[buf][row1 * LDSB + kc1] =
            *(const bf16x8*)(Bm + (size_t)(n0 + row1) * ldb + k0 + kc1);
#endif
    };

    const int nk = K / 32;
    stage(0, 0);
    for (int kt = 0; kt < nk; ++kt) {
        const int  buf       = kt & 1;
        const bool have_next = (kt + 1 < nk);
        if (have_next) stage((kt + 1) * 32, buf ^ 1);
        if (kt + 2 < nk) {  // warm L2 for the slab after next (global_prefetch_b8)
            __builtin_prefetch(A + (size_t)(m0 + row0) * lda + (kt + 2) * 32 + kc0, 0, 1);
            __builtin_prefetch(Bm + (size_t)(n0 + row0) * ldb + (kt + 2) * 32 + kc0, 0, 1);
        }
        // async completes in order: <=4 outstanding => slab `kt` has landed
        if (have_next) wait_async<4>(); else wait_async<0>();
        __syncthreads();

        bf16x16 af[2], bfr[4];
#pragma unroll
        for (int mf = 0; mf < 2; ++mf)
            af[mf] = load_frag(&As[buf][(wm * 32 + mf * 16) * LDSB], LDSB, lane);
#pragma unroll
        for (int nf = 0; nf < 4; ++nf)
            bfr[nf] = load_frag(&Bs[buf][(wn * 64 + nf * 16) * LDSB], LDSB, lane);
#pragma unroll
        for (int mf = 0; mf < 2; ++mf)
#pragma unroll
            for (int nf = 0; nf < 4; ++nf)
                acc[mf][nf] = wmma_bf16(af[mf], bfr[nf], acc[mf][nf]);
        __syncthreads();
    }

    // ---- epilogue ----
    const int r  = lane & 15;
    const int hi = lane >> 4;
#pragma unroll
    for (int mf = 0; mf < 2; ++mf) {
#pragma unroll
        for (int nf = 0; nf < 4; ++nf) {
#pragma unroll
            for (int v = 0; v < 8; ++v) {
                float val    = acc[mf][nf][v];
                const int mg = m0 + wm * 32 + mf * 16 + v + hi * 8;   // row in [B*S]
                const int ng = n0 + wn * 64 + nf * 16 + r;            // col in [D]
                if (MODE == 0) {
                    Cf[(size_t)mg * ldc + ng] = val;
                } else {
                    const int b = mg >> 11, s = mg & 2047;
                    const int h = ng >> 7,  d = ng & 127;
                    if (MODE == 1) {
                        // RoPE: even/odd dims sit in adjacent lanes -> shfl_xor
                        float partner = __shfl_xor(val, 1, 32);
                        float pos = (float)tpos[s];
                        // inv_freq = theta^{-(d/2)/64}, ln(1e4)=9.210340372
                        float inv = __expf(-9.210340372f * (float)(d >> 1) * (1.0f / 64.0f));
                        float sn, cs;
                        __sincosf(pos * inv, &sn, &cs);
                        float outv = val * cs + (((d & 1) == 0) ? -partner : partner) * sn;
                        Obf[(size_t)((b * NHEADS + h) * S_LEN + s) * DK + d] = (__bf16)outv;
                    } else {  // MODE == 2 : V transposed [B][H][Dk][S]
                        Obf[(size_t)((b * NHEADS + h) * DK + d) * S_LEN + s] = (__bf16)val;
                    }
                }
            }
        }
    }
}

// ---------------------------------------------------------------------------
// Flash attention: grid (S/64, B*H), block 256 (8 waves).
// Each wave: 16 q-rows x 64 dims, fully independent key loop (Nk = 64).
// ---------------------------------------------------------------------------
#define PLDS 72   // padded P row stride (halfwords): 144B, 16B aligned

__global__ __launch_bounds__(256)
void flash_attn_kernel(const __bf16* __restrict__ Q,
                       const __bf16* __restrict__ Kt,
                       const __bf16* __restrict__ Vt,
                       __bf16* __restrict__ O) {
    __shared__ __bf16 P[8][16 * PLDS];

    const int t    = threadIdx.x;
    const int lane = t & 31;
    const int wave = t >> 5;
    const int wm   = wave >> 1;   // 0..3 : q-row strip
    const int wn   = wave & 1;    // 0..1 : dim half
    const int q0   = blockIdx.x * 64;
    const int bh   = blockIdx.y;  // b*16 + h
    const int r    = lane & 15;
    const int hi   = lane >> 4;

    const __bf16* Qh = Q  + (size_t)bh * S_LEN * DK;
    const __bf16* Kh = Kt + (size_t)bh * S_LEN * DK;
    const __bf16* Vh = Vt + (size_t)bh * DK * S_LEN;
    const int qrow = q0 + wm * 16;

    // hoist Q fragments (4 x 32 chunks over Dk=128) out of the key loop
    bf16x16 qf[4];
#pragma unroll
    for (int ks = 0; ks < 4; ++ks)
        qf[ks] = load_frag(Qh + (size_t)qrow * DK + ks * 32, DK, lane);

    f32x8 acc[4];
#pragma unroll
    for (int nf = 0; nf < 4; ++nf) acc[nf] = (f32x8){};
    float mrun[8], lrun[8];
#pragma unroll
    for (int v = 0; v < 8; ++v) { mrun[v] = -3.0e38f; lrun[v] = 0.0f; }

    const float scale = 0.08838834764831845f;  // 1/sqrt(128)
    __bf16* myP = &P[wave][0];

    for (int kb = 0; kb <= q0; kb += 64) {
        // ---- scores S = Q * K^T for 16 q-rows x 64 keys ----
        f32x8 sacc[4];
#pragma unroll
        for (int nf = 0; nf < 4; ++nf) sacc[nf] = (f32x8){};
#pragma unroll
        for (int ks = 0; ks < 4; ++ks) {
#pragma unroll
            for (int nf = 0; nf < 4; ++nf) {
                bf16x16 kf = load_frag(Kh + (size_t)(kb + nf * 16) * DK + ks * 32, DK, lane);
                sacc[nf] = wmma_bf16(qf[ks], kf, sacc[nf]);
            }
        }

        // ---- online softmax, per q-row (row = v + 8*hi) ----
        const bool needmask = (kb + 63 > qrow);
#pragma unroll
        for (int v = 0; v < 8; ++v) {
            const int mg = qrow + v + hi * 8;
            float x[4];
#pragma unroll
            for (int nf = 0; nf < 4; ++nf) {
                float xv = sacc[nf][v] * scale;
                const int key = kb + nf * 16 + r;
                x[nf] = (needmask && key > mg) ? -3.0e38f : xv;
            }
            float tmax = fmaxf(fmaxf(x[0], x[1]), fmaxf(x[2], x[3]));
#pragma unroll
            for (int off = 8; off; off >>= 1)
                tmax = fmaxf(tmax, __shfl_xor(tmax, off, 32));
            const float mnew = fmaxf(mrun[v], tmax);
            const float rsc  = __expf(mrun[v] - mnew);
            mrun[v] = mnew;
            float psum = 0.0f;
#pragma unroll
            for (int nf = 0; nf < 4; ++nf) {
                float p = __expf(x[nf] - mnew);
                psum += p;
                myP[(v + 8 * hi) * PLDS + nf * 16 + r] = (__bf16)p;
            }
#pragma unroll
            for (int off = 8; off; off >>= 1)
                psum += __shfl_xor(psum, off, 32);
            lrun[v] = lrun[v] * rsc + psum;
#pragma unroll
            for (int nf = 0; nf < 4; ++nf) acc[nf][v] *= rsc;
        }

        // LDS is in-order within a wave; the wait + memory clobber stops the
        // compiler from hoisting the transposed reads above the stores.
        asm volatile("s_wait_dscnt 0" ::: "memory");

        // ---- O += P * V  (P read back from LDS in A-frag layout) ----
#pragma unroll
        for (int kk = 0; kk < 2; ++kk) {
            bf16x16 pf = load_frag(myP + kk * 32, PLDS, lane);
#pragma unroll
            for (int nf = 0; nf < 4; ++nf) {
                bf16x16 vf = load_frag(
                    Vh + (size_t)(wn * 64 + nf * 16) * S_LEN + kb + kk * 32, S_LEN, lane);
                acc[nf] = wmma_bf16(pf, vf, acc[nf]);
            }
        }
        asm volatile("s_wait_dscnt 0" ::: "memory");
    }

    // ---- normalize and pack to [B][S][H*Dk] bf16 for the output GEMM ----
    const int b = bh >> 4, h = bh & 15;
#pragma unroll
    for (int nf = 0; nf < 4; ++nf) {
#pragma unroll
        for (int v = 0; v < 8; ++v) {
            const int s = qrow + v + 8 * hi;
            const int e = h * DK + wn * 64 + nf * 16 + r;
            float o = acc[nf][v] / lrun[v];
            O[(size_t)(b * S_LEN + s) * DMODEL + e] = (__bf16)o;
        }
    }
}

// ---------------------------------------------------------------------------
// host launch
// ---------------------------------------------------------------------------
extern "C" void kernel_launch(void* const* d_in, const int* in_sizes, int n_in,
                              void* d_out, int out_size, void* d_ws, size_t ws_size,
                              hipStream_t stream) {
    const float* x   = (const float*)d_in[0];
    const int*   tp  = (const int*)d_in[1];
    const float* q_w = (const float*)d_in[2];
    const float* k_w = (const float*)d_in[3];
    const float* v_w = (const float*)d_in[4];
    const float* o_w = (const float*)d_in[5];
    float* out = (float*)d_out;

    const size_t MB = 1024ull * 1024ull;
    char* ws = (char*)d_ws;
    __bf16* Xb  = (__bf16*)(ws + 0);        // 32 MB : x bf16 [B*S][D]
    __bf16* Wqb = (__bf16*)(ws + 32 * MB);  //  8 MB
    __bf16* Wkb = (__bf16*)(ws + 40 * MB);  //  8 MB
    __bf16* Wvb = (__bf16*)(ws + 48 * MB);  //  8 MB
    __bf16* Wob = (__bf16*)(ws + 56 * MB);  //  8 MB
    __bf16* Qb  = (__bf16*)(ws + 64 * MB);  // 32 MB : [B][H][S][Dk]
    __bf16* Kb  = (__bf16*)(ws + 96 * MB);  // 32 MB : [B][H][S][Dk]
    __bf16* Vt  = (__bf16*)(ws + 128 * MB); // 32 MB : [B][H][Dk][S]
    __bf16* At  = (__bf16*)(ws + 160 * MB); // 32 MB : [B*S][D]  (total 192 MB)

    const int nX = NBATCH * S_LEN * DMODEL;   // 16,777,216
    const int nW = DMODEL * DMODEL;           //  4,194,304

    // 1) fp32 -> bf16 conversions
    cvt_f32_bf16_kernel<<<2048, 256, 0, stream>>>(x,   Xb,  nX / 4);
    cvt_f32_bf16_kernel<<<1024, 256, 0, stream>>>(q_w, Wqb, nW / 4);
    cvt_f32_bf16_kernel<<<1024, 256, 0, stream>>>(k_w, Wkb, nW / 4);
    cvt_f32_bf16_kernel<<<1024, 256, 0, stream>>>(v_w, Wvb, nW / 4);
    cvt_f32_bf16_kernel<<<1024, 256, 0, stream>>>(o_w, Wob, nW / 4);

    // 2) projections (M=8192, N=2048, K=2048), fused RoPE / V-transpose
    dim3 ggrid(DMODEL / 128, (NBATCH * S_LEN) / 128);  // (16, 64)
    gemm_bf16_abt_kernel<1><<<ggrid, 256, 0, stream>>>(
        Xb, DMODEL, Wqb, DMODEL, nullptr, 0, Qb, tp, DMODEL);
    gemm_bf16_abt_kernel<1><<<ggrid, 256, 0, stream>>>(
        Xb, DMODEL, Wkb, DMODEL, nullptr, 0, Kb, tp, DMODEL);
    gemm_bf16_abt_kernel<2><<<ggrid, 256, 0, stream>>>(
        Xb, DMODEL, Wvb, DMODEL, nullptr, 0, Vt, tp, DMODEL);

    // 3) causal flash attention
    dim3 fgrid(S_LEN / 64, NBATCH * NHEADS);           // (32, 64)
    flash_attn_kernel<<<fgrid, 256, 0, stream>>>(Qb, Kb, Vt, At);

    // 4) output projection -> fp32 d_out
    gemm_bf16_abt_kernel<0><<<ggrid, 256, 0, stream>>>(
        At, DMODEL, Wob, DMODEL, out, DMODEL, nullptr, nullptr, DMODEL);
}
